// Block_68341519614809
// MI455X (gfx1250) — compile-verified, matrix operand order checked by reading, not run on hardware
//
#include <hip/hip_runtime.h>
#include <hip/hip_bf16.h>

// ---------------- problem constants ----------------
#define BB 4
#define TT 2048
#define DD 1024
#define HH 4
#define DH 256
#define BT (BB * TT)           // 8192
#define CONVK 4
#define VSPLIT 8               // delta-rule v-column split
#define VCOLS (DH / VSPLIT)    // 32 columns per block
#define SPAD 33                // LDS row stride for S (bank decorrelation)

typedef __bf16  v16bf __attribute__((ext_vector_type(16)));
typedef float   v8f   __attribute__((ext_vector_type(8)));

// fp32 -> bf16 (round to nearest even)
__device__ __forceinline__ unsigned short f2bf(float f) {
    unsigned int u = __float_as_uint(f);
    u = (u + 0x7FFFu + ((u >> 16) & 1u)) >> 16;
    return (unsigned short)u;
}

__device__ __forceinline__ float sigmoidf_(float x) {
    return 1.0f / (1.0f + __expf(-x));
}

// ---------------- elementwise converts ----------------
__global__ void conv_f32_to_bf16(const float* __restrict__ in,
                                 unsigned short* __restrict__ out, int n) {
    int i = blockIdx.x * blockDim.x + threadIdx.x;
    if (i < n) out[i] = f2bf(in[i]);
}

// Wt[n*K + k] = bf16(W[k*N + n]) so the B operand streams contiguously in K
__global__ void transpose_to_bf16(const float* __restrict__ W,
                                  unsigned short* __restrict__ Wt,
                                  int K, int N) {
    int i = blockIdx.x * blockDim.x + threadIdx.x;
    if (i >= K * N) return;
    int n = i / K;
    int k = i - n * K;
    Wt[i] = f2bf(W[(size_t)k * N + n]);
}

// ---------------- WMMA bf16 GEMM: C[M,N] = A[M,K] * B[K,N] ----------------
// A row-major bf16; Bt is B transposed (N-major, K contiguous) bf16; C fp32.
// Workgroup 256 thr = 8 waves; per-wave 32x32 tile; block tile 128x64.
union FragU { uint4 q[2]; v16bf v; };

__device__ __forceinline__ v16bf load_frag16(const unsigned short* p) {
    // lane holds two contiguous 8-element (16B) chunks: K {base..base+7} and {base+16..base+23}
    FragU f;
    f.q[0] = *reinterpret_cast<const uint4*>(p);
    f.q[1] = *reinterpret_cast<const uint4*>(p + 16);
    return f.v;
}

__global__ __launch_bounds__(256)
void gemm_bf16_wmma(const unsigned short* __restrict__ A,
                    const unsigned short* __restrict__ Bt,
                    float* __restrict__ C, int M, int N, int K) {
    const int tid  = threadIdx.x;
    const int wave = tid >> 5;
    const int lane = tid & 31;
    const int wm = wave & 3;       // 4 row groups
    const int wn = wave >> 2;      // 2 col groups
    const int lrow = lane & 15;    // row/col within 16
    const int kg   = lane >> 4;    // K half select

    const int rowBase = blockIdx.y * 128 + wm * 32;
    const int colBase = blockIdx.x * 64  + wn * 32;

    v8f acc0 = {}, acc1 = {}, acc2 = {}, acc3 = {};

    const unsigned short* a0p = A  + (size_t)(rowBase      + lrow) * K + kg * 8;
    const unsigned short* a1p = A  + (size_t)(rowBase + 16 + lrow) * K + kg * 8;
    const unsigned short* b0p = Bt + (size_t)(colBase      + lrow) * K + kg * 8;
    const unsigned short* b1p = Bt + (size_t)(colBase + 16 + lrow) * K + kg * 8;

    for (int kb = 0; kb < K; kb += 32) {
        __builtin_prefetch(a0p + kb + 128, 0, 1);
        __builtin_prefetch(b0p + kb + 128, 0, 1);
        v16bf a0 = load_frag16(a0p + kb);
        v16bf a1 = load_frag16(a1p + kb);
        v16bf b0 = load_frag16(b0p + kb);
        v16bf b1 = load_frag16(b1p + kb);
        acc0 = __builtin_amdgcn_wmma_f32_16x16x32_bf16(false, a0, false, b0, (short)0, acc0, false, false);
        acc1 = __builtin_amdgcn_wmma_f32_16x16x32_bf16(false, a0, false, b1, (short)0, acc1, false, false);
        acc2 = __builtin_amdgcn_wmma_f32_16x16x32_bf16(false, a1, false, b0, (short)0, acc2, false, false);
        acc3 = __builtin_amdgcn_wmma_f32_16x16x32_bf16(false, a1, false, b1, (short)0, acc3, false, false);
    }

    // D layout: VGPR r -> (M = r + kg*8, N = lrow) within 16x16 tile
    v8f accs[4] = {acc0, acc1, acc2, acc3};
    #pragma unroll
    for (int i = 0; i < 4; ++i) {
        const int rb = rowBase + (i >> 1) * 16 + kg * 8;
        const int cb = colBase + (i & 1) * 16 + lrow;
        #pragma unroll
        for (int r = 0; r < 8; ++r)
            C[(size_t)(rb + r) * N + cb] = accs[i][r];
    }
}

// ---------------- beta = sigmoid(x @ Wb), Wb: (D, H=4) ----------------
__global__ __launch_bounds__(256)
void beta_kernel(const float* __restrict__ x, const float* __restrict__ Wb,
                 float* __restrict__ Beta) {
    __shared__ float red[HH][256];
    const int m = blockIdx.x;
    const int tid = threadIdx.x;
    const float* xr = x + (size_t)m * DD;
    float a0 = 0.f, a1 = 0.f, a2 = 0.f, a3 = 0.f;
    for (int k = tid; k < DD; k += 256) {
        float xv = xr[k];
        const float* w = Wb + (size_t)k * HH;
        a0 += xv * w[0]; a1 += xv * w[1]; a2 += xv * w[2]; a3 += xv * w[3];
    }
    red[0][tid] = a0; red[1][tid] = a1; red[2][tid] = a2; red[3][tid] = a3;
    __syncthreads();
    for (int s = 128; s > 0; s >>= 1) {
        if (tid < s) {
            #pragma unroll
            for (int h = 0; h < HH; ++h) red[h][tid] += red[h][tid + s];
        }
        __syncthreads();
    }
    if (tid < HH) Beta[(size_t)m * HH + tid] = sigmoidf_(red[tid][0]);
}

// ---------------- depthwise causal conv (K=4) + SiLU ----------------
__global__ void conv_silu_kernel(const float* __restrict__ U,
                                 const float* __restrict__ W,
                                 float* __restrict__ Out, int total) {
    int idx = blockIdx.x * blockDim.x + threadIdx.x;
    if (idx >= total) return;
    int d = idx & (DD - 1);
    int t = (idx / DD) & (TT - 1);
    float s = 0.f;
    #pragma unroll
    for (int j = 0; j < CONVK; ++j) {
        int dt = (CONVK - 1) - j;
        if (t >= dt) s += W[j * DD + d] * U[idx - dt * DD];
    }
    Out[idx] = s * sigmoidf_(s);
}

// ---------------- per-head L2 norm (in place), DH == blockDim ----------------
__global__ __launch_bounds__(256)
void l2norm_kernel(float* __restrict__ X) {
    __shared__ float red[256];
    const int tid = threadIdx.x;
    const size_t base = (size_t)blockIdx.x * DH;
    float v = X[base + tid];
    red[tid] = v * v;
    __syncthreads();
    for (int s = 128; s > 0; s >>= 1) {
        if (tid < s) red[tid] += red[tid + s];
        __syncthreads();
    }
    X[base + tid] = v * rsqrtf(red[0] + 1e-6f);
}

// ---------------- delta rule scan ----------------
// grid = (B*H, VSPLIT), block = 256 (8 k-slices x 32 v-cols)
// Columns of S evolve independently -> split v across blocks.
__global__ __launch_bounds__(256)
void delta_rule_kernel(const float* __restrict__ Q, const float* __restrict__ K,
                       const float* __restrict__ V, const float* __restrict__ Beta,
                       float* __restrict__ O) {
    __shared__ float S[DH * SPAD];
    __shared__ float kt[DH], qt[DH], vt[VCOLS], dl[VCOLS];
    __shared__ float red[256];
    __shared__ float betas;

    const int bh = blockIdx.x;            // b*H + h
    const int vbase = blockIdx.y * VCOLS;
    const int tid = threadIdx.x;
    const int ks = tid >> 5;              // k-slice 0..7
    const int v  = tid & 31;              // v column within slice

    for (int i = tid; i < DH * SPAD; i += 256) S[i] = 0.f;
    __syncthreads();

    for (int t = 0; t < TT; ++t) {
        const size_t rowoff = ((size_t)(bh / HH) * TT + t) * (HH * DH) + (size_t)(bh % HH) * DH;
        kt[tid] = K[rowoff + tid];
        qt[tid] = Q[rowoff + tid];
        if (tid < VCOLS) vt[tid] = V[rowoff + vbase + tid];
        if (tid == 0)    betas = Beta[((size_t)(bh / HH) * TT + t) * HH + (bh % HH)];
        __syncthreads();

        // kS[v] partial over this thread's k-slice
        const int k0 = ks * 32;
        float p = 0.f;
        #pragma unroll
        for (int kk = 0; kk < 32; ++kk) p += kt[k0 + kk] * S[(k0 + kk) * SPAD + v];
        red[tid] = p;
        __syncthreads();

        if (ks == 0) {
            float kS = 0.f;
            #pragma unroll
            for (int g = 0; g < VSPLIT; ++g) kS += red[g * 32 + v];
            dl[v] = (vt[v] - kS) * betas;
        }
        __syncthreads();

        // fused: S += outer(k, delta); o = q . S (with updated S)
        const float d = dl[v];
        float po = 0.f;
        #pragma unroll
        for (int kk = 0; kk < 32; ++kk) {
            const int k = k0 + kk;
            float s = S[k * SPAD + v] + kt[k] * d;
            S[k * SPAD + v] = s;
            po += qt[k] * s;
        }
        red[tid] = po;
        __syncthreads();

        if (ks == 0) {
            float o = 0.f;
            #pragma unroll
            for (int g = 0; g < VSPLIT; ++g) o += red[g * 32 + v];
            O[rowoff + vbase + v] = o;
        }
        __syncthreads();   // protect kt/qt/red for next step
    }
}

// ---------------- RMS norm * g_rms, fused with bf16 convert ----------------
__global__ __launch_bounds__(256)
void rmsnorm_bf16_kernel(const float* __restrict__ O, const float* __restrict__ g,
                         unsigned short* __restrict__ Obf) {
    __shared__ float red[256];
    const int tid = threadIdx.x;
    const size_t base = (size_t)blockIdx.x * DH;
    float vv = O[base + tid];
    red[tid] = vv * vv;
    __syncthreads();
    for (int s = 128; s > 0; s >>= 1) {
        if (tid < s) red[tid] += red[tid + s];
        __syncthreads();
    }
    float scale = rsqrtf(red[0] * (1.0f / DH) + 1e-5f);
    Obf[base + tid] = f2bf(vv * scale * g[tid]);
}

// ---------------- residual + LayerNorm ----------------
__global__ __launch_bounds__(256)
void final_ln_kernel(const float* __restrict__ Yg, const float* __restrict__ x,
                     const float* __restrict__ ln_g, const float* __restrict__ ln_b,
                     float* __restrict__ out) {
    __shared__ float rs[256], rq[256];
    const int tid = threadIdx.x;
    const size_t base = (size_t)blockIdx.x * DD;
    float y[4]; float s = 0.f, ss = 0.f;
    #pragma unroll
    for (int i = 0; i < 4; ++i) {
        int c = tid + i * 256;
        y[i] = Yg[base + c] + x[base + c];
        s += y[i]; ss += y[i] * y[i];
    }
    rs[tid] = s; rq[tid] = ss;
    __syncthreads();
    for (int st = 128; st > 0; st >>= 1) {
        if (tid < st) { rs[tid] += rs[tid + st]; rq[tid] += rq[tid + st]; }
        __syncthreads();
    }
    float mu  = rs[0] * (1.0f / DD);
    float var = rq[0] * (1.0f / DD) - mu * mu;
    float inv = rsqrtf(var + 1e-5f);
    #pragma unroll
    for (int i = 0; i < 4; ++i) {
        int c = tid + i * 256;
        out[base + c] = (y[i] - mu) * inv * ln_g[c] + ln_b[c];
    }
}

// ---------------- orchestration ----------------
extern "C" void kernel_launch(void* const* d_in, const int* in_sizes, int n_in,
                              void* d_out, int out_size, void* d_ws, size_t ws_size,
                              hipStream_t stream) {
    const float* x      = (const float*)d_in[0];
    const float* Wq     = (const float*)d_in[1];
    const float* Wk     = (const float*)d_in[2];
    const float* Wv     = (const float*)d_in[3];
    const float* Wb     = (const float*)d_in[4];
    const float* conv_q = (const float*)d_in[5];
    const float* conv_k = (const float*)d_in[6];
    const float* conv_v = (const float*)d_in[7];
    const float* g_rms  = (const float*)d_in[8];
    const float* Wo     = (const float*)d_in[9];
    const float* ln_g   = (const float*)d_in[10];
    const float* ln_b   = (const float*)d_in[11];
    float* out = (float*)d_out;

    // workspace carve
    char* base = (char*)d_ws;
    size_t off = 0;
    auto carve = [&](size_t bytes) -> void* {
        void* p = base + off;
        off += (bytes + 255) & ~(size_t)255;
        return p;
    };
    unsigned short* Xb  = (unsigned short*)carve((size_t)BT * DD * 2);
    unsigned short* Wqt = (unsigned short*)carve((size_t)DD * DD * 2);
    unsigned short* Wkt = (unsigned short*)carve((size_t)DD * DD * 2);
    unsigned short* Wvt = (unsigned short*)carve((size_t)DD * DD * 2);
    unsigned short* Wot = (unsigned short*)carve((size_t)DD * DD * 2);
    float* Qf   = (float*)carve((size_t)BT * DD * 4);
    float* Kf   = (float*)carve((size_t)BT * DD * 4);
    float* Vf   = (float*)carve((size_t)BT * DD * 4);
    float* QC   = (float*)carve((size_t)BT * DD * 4);
    float* KC   = (float*)carve((size_t)BT * DD * 4);
    float* VC   = (float*)carve((size_t)BT * DD * 4);
    float* BetaB = (float*)carve((size_t)BT * HH * 4);
    // buffer reuse after producers are dead
    float* O            = Qf;                     // delta-rule output
    unsigned short* Obf = (unsigned short*)Kf;    // bf16 of normed O
    float* Yg           = Vf;                     // O @ Wo

    const int totalXD = BT * DD;                  // 8,388,608
    const int ew_blocks = (totalXD + 255) / 256;
    const int w_blocks  = (DD * DD + 255) / 256;

    // 1) converts
    conv_f32_to_bf16<<<ew_blocks, 256, 0, stream>>>(x, Xb, totalXD);
    transpose_to_bf16<<<w_blocks, 256, 0, stream>>>(Wq, Wqt, DD, DD);
    transpose_to_bf16<<<w_blocks, 256, 0, stream>>>(Wk, Wkt, DD, DD);
    transpose_to_bf16<<<w_blocks, 256, 0, stream>>>(Wv, Wvt, DD, DD);
    transpose_to_bf16<<<w_blocks, 256, 0, stream>>>(Wo, Wot, DD, DD);

    // 2) q/k/v projections (WMMA)
    dim3 ggrid(DD / 64, BT / 128);
    gemm_bf16_wmma<<<ggrid, 256, 0, stream>>>(Xb, Wqt, Qf, BT, DD, DD);
    gemm_bf16_wmma<<<ggrid, 256, 0, stream>>>(Xb, Wkt, Kf, BT, DD, DD);
    gemm_bf16_wmma<<<ggrid, 256, 0, stream>>>(Xb, Wvt, Vf, BT, DD, DD);

    // 3) beta
    beta_kernel<<<BT, 256, 0, stream>>>(x, Wb, BetaB);

    // 4) depthwise causal conv + SiLU
    conv_silu_kernel<<<ew_blocks, 256, 0, stream>>>(Qf, conv_q, QC, totalXD);
    conv_silu_kernel<<<ew_blocks, 256, 0, stream>>>(Kf, conv_k, KC, totalXD);
    conv_silu_kernel<<<ew_blocks, 256, 0, stream>>>(Vf, conv_v, VC, totalXD);

    // 5) L2 norm per head on q, k
    l2norm_kernel<<<BT * HH, 256, 0, stream>>>(QC);
    l2norm_kernel<<<BT * HH, 256, 0, stream>>>(KC);

    // 6) delta-rule scan (v-columns split across blocks)
    delta_rule_kernel<<<dim3(BB * HH, VSPLIT), 256, 0, stream>>>(QC, KC, VC, BetaB, O);

    // 7) RMS norm * g_rms -> bf16
    rmsnorm_bf16_kernel<<<BT * HH, 256, 0, stream>>>(O, g_rms, Obf);

    // 8) output projection (WMMA)
    gemm_bf16_wmma<<<ggrid, 256, 0, stream>>>(Obf, Wot, Yg, BT, DD, DD);

    // 9) residual + LayerNorm
    final_ln_kernel<<<BT, 256, 0, stream>>>(Yg, x, ln_g, ln_b, out);
}
